// SelectiveCausalAttention_16466904613463
// MI455X (gfx1250) — matrix operand order repaired and checked by reading, not compiled, not run on hardware
//
#include <hip/hip_runtime.h>
#include <hip/hip_bf16.h>

typedef __attribute__((ext_vector_type(16))) _Float16 v16h;
typedef __attribute__((ext_vector_type(8)))  _Float16 v8h;
typedef __attribute__((ext_vector_type(8)))  float    v8f;

#define WPB 8  // waves per block (256 threads, wave32)

// ---------------------------------------------------------------------------
// Fragment loaders, contiguous k order, same for A and B (16-bit A 16x32, B 32x16, f32 C/D)
// A: lanes 0-15 hold K{0..15}, lanes 16-31 hold K{16..31}
//    -> per-lane: 16 contiguous halves            (p = row + k0 + (lane>>4)*16)
// B: lanes 0-15 hold K{0..15}, lanes 16-31 hold K{16..31}
//    -> per-lane: 16 contiguous halves            (p = col + k0 + (lane>>4)*16)
// ---------------------------------------------------------------------------
__device__ __forceinline__ v16h load_frag_pair(const _Float16* p, int hiOff) {
    v8h lo = *(const v8h*)p;
    v8h hi = *(const v8h*)(p + hiOff);
    v16h r;
#pragma unroll
    for (int i = 0; i < 8; ++i) { r[i] = lo[i]; r[i + 8] = hi[i]; }
    return r;
}

// ---------------------------------------------------------------------------
// Generic batched GEMM:  C[b] = alpha * A[b] @ Bt[b]^T + bias
//   A : f16 [M,K]   row-major, leading dim ldA, batch stride sA
//   Bt: f16 [N,K]   (B transposed), leading dim ldB, batch stride sB
//   C : f32 [M,N]   leading dim ldC, batch stride sC
// One wave computes a 16x64 strip: A fragment reused across 4 WMMAs per
// K-step (4 v_wmma_f32_16x16x32_f16 per 10 b128 loads).
// M mult of 16, N mult of 64, K mult of 32.
// ---------------------------------------------------------------------------
__global__ __launch_bounds__(256) void gemm_wmma_kernel(
    const _Float16* __restrict__ A, const _Float16* __restrict__ Bt,
    float* __restrict__ C, const float* __restrict__ bias, float alpha,
    int N, int K, int ldA, int ldB, int ldC,
    long long sA, long long sB, long long sC,
    int tilesPerBatch, int totalWaves)
{
    const int wave = threadIdx.x >> 5;
    const int lane = threadIdx.x & 31;
    const int gid  = blockIdx.x * WPB + wave;
    if (gid >= totalWaves) return;            // wave-uniform: EXEC stays all-1s

    const int tilesN = N >> 6;                // 64-wide strips
    const int batch  = gid / tilesPerBatch;
    const int t      = gid - batch * tilesPerBatch;
    const int tm     = t / tilesN;
    const int tn     = t - tm * tilesN;

    const int lm = lane & 15;
    const int kh = lane >> 4;

    const _Float16* pA = A + (long long)batch * sA
                           + (long long)(tm * 16 + lm) * ldA + kh * 16;
    const _Float16* pB = Bt + (long long)batch * sB
                           + (long long)(tn * 64 + lm) * ldB + kh * 16;
    const long long ldB16 = (long long)ldB * 16;   // next 16 columns of B

    v8f acc0 = {}, acc1 = {}, acc2 = {}, acc3 = {};
    for (int k0 = 0; k0 < K; k0 += 32) {
        v16h af = load_frag_pair(pA, 8);
        v16h b0 = load_frag_pair(pB, 8);
        v16h b1 = load_frag_pair(pB + ldB16, 8);
        v16h b2 = load_frag_pair(pB + 2 * ldB16, 8);
        v16h b3 = load_frag_pair(pB + 3 * ldB16, 8);
        acc0 = __builtin_amdgcn_wmma_f32_16x16x32_f16(
            false, af, false, b0, (short)0, acc0, false, false);
        acc1 = __builtin_amdgcn_wmma_f32_16x16x32_f16(
            false, af, false, b1, (short)0, acc1, false, false);
        acc2 = __builtin_amdgcn_wmma_f32_16x16x32_f16(
            false, af, false, b2, (short)0, acc2, false, false);
        acc3 = __builtin_amdgcn_wmma_f32_16x16x32_f16(
            false, af, false, b3, (short)0, acc3, false, false);
        pA += 32;
        pB += 32;
        // gfx1250 prefetch of the next K panel (global_prefetch_b8)
        __builtin_prefetch(pA, 0, 1);
        __builtin_prefetch(pB, 0, 1);
    }

    float* pC = C + (long long)batch * sC
                  + (long long)(tm * 16 + (kh << 3)) * ldC + tn * 64 + lm;
#define EPILOGUE(ACC, S)                                                      \
    {                                                                         \
        const float bv = bias ? bias[tn * 64 + (S) * 16 + lm] : 0.0f;         \
        _Pragma("unroll") for (int r = 0; r < 8; ++r)                         \
            pC[(long long)r * ldC + (S) * 16] = alpha * ACC[r] + bv;          \
    }
    EPILOGUE(acc0, 0)
    EPILOGUE(acc1, 1)
    EPILOGUE(acc2, 2)
    EPILOGUE(acc3, 3)
#undef EPILOGUE
}

// ---------------------------------------------------------------------------
// Elementwise f32 -> f16
// ---------------------------------------------------------------------------
__global__ void cvt_f16_kernel(const float* __restrict__ s,
                               _Float16* __restrict__ d, int n) {
    int i = blockIdx.x * 256 + threadIdx.x;
    if (i < n) d[i] = (_Float16)s[i];
}

// ---------------------------------------------------------------------------
// Transpose+convert: src f32 [K,N] -> dst f16 [N,K]
// ---------------------------------------------------------------------------
__global__ void pack_bt_kernel(const float* __restrict__ s,
                               _Float16* __restrict__ d, int K, int N) {
    int i = blockIdx.x * 256 + threadIdx.x;
    if (i >= K * N) return;
    int k = i / N, n = i - k * N;
    d[(long long)n * K + k] = (_Float16)s[i];
}

// ---------------------------------------------------------------------------
// Split qkv f32 [B,L,3,H,64] -> q16,k16 f16 [B,H,L,64], vt16 f16 [B,H,64,L]
// ---------------------------------------------------------------------------
__global__ void split_qkv_kernel(const float* __restrict__ qkv,
                                 _Float16* __restrict__ q16,
                                 _Float16* __restrict__ k16,
                                 _Float16* __restrict__ vt16, int total) {
    int i = blockIdx.x * 256 + threadIdx.x;
    if (i >= total) return;
    int di = i & 63;
    int l  = (i >> 6) & 511;
    int h  = (i >> 15) % 12;
    int b  = i / (64 * 512 * 12);
    const float* base = qkv + (long long)((b * 512 + l) * 3) * 768 + h * 64 + di;
    int bh = b * 12 + h;
    long long qi = ((long long)bh * 512 + l) * 64 + di;
    q16[qi] = (_Float16)base[0];
    k16[qi] = (_Float16)base[768];
    vt16[((long long)bh * 64 + di) * 512 + l] = (_Float16)base[1536];
}

// ---------------------------------------------------------------------------
// ctx f32 [B,H,L,64] -> ctx16 f16 [B,L,768]
// ---------------------------------------------------------------------------
__global__ void ctx_pack_kernel(const float* __restrict__ ctx,
                                _Float16* __restrict__ d, int total) {
    int i = blockIdx.x * 256 + threadIdx.x;
    if (i >= total) return;
    int di = i & 63;
    int l  = (i >> 6) & 511;
    int h  = (i >> 15) % 12;
    int b  = i / (64 * 512 * 12);
    float v = ctx[((long long)(b * 12 + h) * 512 + l) * 64 + di];
    d[(long long)(b * 512 + l) * 768 + h * 64 + di] = (_Float16)v;
}

// ---------------------------------------------------------------------------
// Fused pairwise gate + bias + softmax.
// One block per (b,h,i) row. a already includes b1 (folded as GEMM bias).
//   s_j = scores[j] + gamma[h]*A_c[i,j]*sigmoid(sum_d relu(a_d + c_jd)*w2_d + b2)
//   attn[j] = softmax_j(s)  (f16 out)
// ---------------------------------------------------------------------------
__global__ __launch_bounds__(256) void gate_softmax_kernel(
    const float* __restrict__ scores, const float* __restrict__ a,
    const float* __restrict__ c, const float* __restrict__ Ac,
    const float* __restrict__ w2, const float* __restrict__ b2,
    const float* __restrict__ gamma, _Float16* __restrict__ attn)
{
    __shared__ float sa[64], sw[64], srow[512], red[256];
    const int tid = threadIdx.x;
    const int bid = blockIdx.x;          // = (b*12 + h)*512 + i
    const int i   = bid & 511;
    const int h   = (bid >> 9) % 12;
    const int b   = bid / (512 * 12);
    const int bh  = b * 12 + h;

    if (tid < 64) {
        sa[tid] = a[((long long)bh * 512 + i) * 64 + tid];
        sw[tid] = w2[tid];
    }
    __syncthreads();

    const float gm  = gamma[h];
    const float b2v = b2[0];
    const float* crow = c + (long long)bh * 512 * 64;
    const float* sc   = scores + ((long long)bh * 512 + i) * 512;
    const float* acr  = Ac + ((long long)b * 512 + i) * 512;

    float lmax = -3.0e38f;
    for (int j = tid; j < 512; j += 256) {
        const float4* cj = (const float4*)(crow + (long long)j * 64);
        float acc = 0.f;
#pragma unroll
        for (int dd = 0; dd < 16; ++dd) {
            float4 cv = cj[dd];
            float v0 = fmaxf(sa[4 * dd + 0] + cv.x, 0.f);
            float v1 = fmaxf(sa[4 * dd + 1] + cv.y, 0.f);
            float v2 = fmaxf(sa[4 * dd + 2] + cv.z, 0.f);
            float v3 = fmaxf(sa[4 * dd + 3] + cv.w, 0.f);
            acc = fmaf(v0, sw[4 * dd + 0], acc);
            acc = fmaf(v1, sw[4 * dd + 1], acc);
            acc = fmaf(v2, sw[4 * dd + 2], acc);
            acc = fmaf(v3, sw[4 * dd + 3], acc);
        }
        float g = 1.f / (1.f + __expf(-(acc + b2v)));
        float s = sc[j] + gm * acr[j] * g;
        srow[j] = s;
        lmax = fmaxf(lmax, s);
    }
    red[tid] = lmax;
    __syncthreads();
    for (int st = 128; st > 0; st >>= 1) {
        if (tid < st) red[tid] = fmaxf(red[tid], red[tid + st]);
        __syncthreads();
    }
    const float mx = red[0];
    __syncthreads();

    float lsum = 0.f;
    for (int j = tid; j < 512; j += 256) {
        float e = __expf(srow[j] - mx);
        srow[j] = e;
        lsum += e;
    }
    red[tid] = lsum;
    __syncthreads();
    for (int st = 128; st > 0; st >>= 1) {
        if (tid < st) red[tid] += red[tid + st];
        __syncthreads();
    }
    const float inv = 1.f / red[0];
    _Float16* arow = attn + ((long long)bh * 512 + i) * 512;
    for (int j = tid; j < 512; j += 256)
        arow[j] = (_Float16)(srow[j] * inv);
}

// ---------------------------------------------------------------------------
// Host side
// ---------------------------------------------------------------------------
static inline int cdiv(long long a, long long b) { return (int)((a + b - 1) / b); }

extern "C" void kernel_launch(void* const* d_in, const int* in_sizes, int n_in,
                              void* d_out, int out_size, void* d_ws, size_t ws_size,
                              hipStream_t stream) {
    const float* x     = (const float*)d_in[0];
    const float* Ac    = (const float*)d_in[1];
    const float* Wqkv  = (const float*)d_in[2];
    const float* bqkv  = (const float*)d_in[3];
    const float* W1q   = (const float*)d_in[4];
    const float* W1k   = (const float*)d_in[5];
    const float* b1    = (const float*)d_in[6];
    const float* w2    = (const float*)d_in[7];
    const float* b2    = (const float*)d_in[8];
    const float* gamma = (const float*)d_in[9];
    const float* Wout  = (const float*)d_in[10];
    const float* bout  = (const float*)d_in[11];
    float* out = (float*)d_out;

    const int B = 2, L = 512, D = 768, H = 12, HD = 64;
    const int BH = B * H;            // 24
    const int BL = B * L;            // 1024
    const int D3 = 3 * D;            // 2304
    const long long NQ  = (long long)BH * L * HD;   // 786432
    const long long NS  = (long long)BH * L * L;    // 6291456

    // workspace carve-up (256B aligned)
    char* ws = (char*)d_ws;
    size_t off = 0;
    auto carve = [&](size_t bytes) {
        size_t o = off;
        off = (off + bytes + 255) & ~(size_t)255;
        return (void*)(ws + o);
    };
    _Float16* x16    = (_Float16*)carve((size_t)BL * D * 2);
    _Float16* wqkvt  = (_Float16*)carve((size_t)D3 * D * 2);
    float*    qkv    = (float*)   carve((size_t)BL * D3 * 4);
    _Float16* q16    = (_Float16*)carve((size_t)NQ * 2);
    _Float16* k16    = (_Float16*)carve((size_t)NQ * 2);
    _Float16* vt16   = (_Float16*)carve((size_t)NQ * 2);
    _Float16* w1qt   = (_Float16*)carve((size_t)HD * HD * 2);
    _Float16* w1kt   = (_Float16*)carve((size_t)HD * HD * 2);
    float*    abuf   = (float*)   carve((size_t)NQ * 4);
    float*    cbuf   = (float*)   carve((size_t)NQ * 4);
    float*    scores = (float*)   carve((size_t)NS * 4);
    _Float16* attn16 = (_Float16*)carve((size_t)NS * 2);
    float*    ctx    = (float*)   carve((size_t)NQ * 4);
    _Float16* ctx16  = (_Float16*)carve((size_t)NQ * 2);
    _Float16* woutt  = (_Float16*)carve((size_t)D * D * 2);
    (void)ws_size;

    // 1) packs / converts
    cvt_f16_kernel<<<cdiv((long long)BL * D, 256), 256, 0, stream>>>(x, x16, BL * D);
    pack_bt_kernel<<<cdiv((long long)D * D3, 256), 256, 0, stream>>>(Wqkv, wqkvt, D, D3);
    pack_bt_kernel<<<cdiv(HD * HD, 256), 256, 0, stream>>>(W1q, w1qt, HD, HD);
    pack_bt_kernel<<<cdiv(HD * HD, 256), 256, 0, stream>>>(W1k, w1kt, HD, HD);
    pack_bt_kernel<<<cdiv((long long)D * D, 256), 256, 0, stream>>>(Wout, woutt, D, D);

    // 2) qkv = x @ Wqkv + bqkv        [1024,768]x[768,2304]
    {
        int tiles = (BL / 16) * (D3 / 64), waves = tiles;
        gemm_wmma_kernel<<<cdiv(waves, WPB), 256, 0, stream>>>(
            x16, wqkvt, qkv, bqkv, 1.0f, D3, D, D, D, D3, 0, 0, 0, tiles, waves);
    }

    // 3) split into q16/k16 [B,H,L,64] and vt16 [B,H,64,L]
    split_qkv_kernel<<<cdiv(NQ, 256), 256, 0, stream>>>(qkv, q16, k16, vt16, (int)NQ);

    // 4) a = q @ W1q + b1 ; c = k @ W1k      (flat [12288,64]x[64,64])
    {
        int M = BH * L;
        int tiles = (M / 16) * (HD / 64), waves = tiles;
        gemm_wmma_kernel<<<cdiv(waves, WPB), 256, 0, stream>>>(
            q16, w1qt, abuf, b1, 1.0f, HD, HD, HD, HD, HD, 0, 0, 0, tiles, waves);
        gemm_wmma_kernel<<<cdiv(waves, WPB), 256, 0, stream>>>(
            k16, w1kt, cbuf, nullptr, 1.0f, HD, HD, HD, HD, HD, 0, 0, 0, tiles, waves);
    }

    // 5) scores = (q @ k^T) * d^-0.5   batched [512,64]x[64,512] x 24
    {
        int tiles = (L / 16) * (L / 64);
        int waves = tiles * BH;
        gemm_wmma_kernel<<<cdiv(waves, WPB), 256, 0, stream>>>(
            q16, k16, scores, nullptr, 0.125f, L, HD, HD, HD, L,
            (long long)L * HD, (long long)L * HD, (long long)L * L, tiles, waves);
    }

    // 6) fused pairwise gate + softmax -> attn16
    gate_softmax_kernel<<<BH * L, 256, 0, stream>>>(
        scores, abuf, cbuf, Ac, w2, b2, gamma, attn16);

    // 7) ctx = attn @ v   batched [512,512]x[512,64] x 24  (Bt = vt16)
    {
        int tiles = (L / 16) * (HD / 64);
        int waves = tiles * BH;
        gemm_wmma_kernel<<<cdiv(waves, WPB), 256, 0, stream>>>(
            attn16, vt16, ctx, nullptr, 1.0f, HD, L, L, L, HD,
            (long long)L * L, (long long)HD * L, (long long)L * HD, tiles, waves);
    }

    // 8) ctx [B,H,L,64] -> ctx16 [B,L,768]
    ctx_pack_kernel<<<cdiv(NQ, 256), 256, 0, stream>>>(ctx, ctx16, (int)NQ);

    // 9) out = ctx @ Wout + bout      [1024,768]x[768,768]
    {
        int tiles = (BL / 16) * (D / 64), waves = tiles;
        gemm_wmma_kernel<<<cdiv(waves, WPB), 256, 0, stream>>>(
            ctx16, woutt, out, bout, 1.0f, D, D, D, D, D, 0, 0, 0, tiles, waves);
    }
}
